// RAWTextureDetector_20942260535487
// MI455X (gfx1250) — matrix operand order repaired and checked by reading, not compiled, not run on hardware
//
#include <hip/hip_runtime.h>
#include <hip/hip_bf16.h>

typedef __attribute__((ext_vector_type(16))) _Float16 v16h;
typedef __attribute__((ext_vector_type(8)))  _Float16 v8h;
typedef __attribute__((ext_vector_type(4)))  _Float16 v4h;
typedef __attribute__((ext_vector_type(8)))  float    v8f;

#if __has_builtin(__builtin_amdgcn_global_load_async_to_lds_b32) && \
    __has_builtin(__builtin_amdgcn_s_wait_asynccnt)
#define HAS_ASYNC_LDS 1
typedef __attribute__((address_space(1))) int* glob_i32p;
typedef __attribute__((address_space(3))) int* lds_i32p;
#else
#define HAS_ASYNC_LDS 0
#endif

// ---------------------------------------------------------------------------
// Pass 0: per-(b,c) statistics -> lower/upper thresholds (32 floats in ws)
// std(x*cw) == cw_c * std(x), so one sum/sumsq reduction suffices.
// ---------------------------------------------------------------------------
__global__ __launch_bounds__(256) void stats_kernel(
    const float* __restrict__ x, const float* __restrict__ chw,
    float* __restrict__ stats)
{
    __shared__ float rs[256], rs2[256];
    const int bc = blockIdx.x;                 // 0..15 == b*4+c
    const float* p = x + (size_t)bc * 1024u * 1024u;
    float s = 0.f, s2 = 0.f;
    for (int i = threadIdx.x; i < 1024 * 1024; i += 256) {
        float v = p[i];
        s += v; s2 += v * v;
    }
    rs[threadIdx.x] = s; rs2[threadIdx.x] = s2;
    __syncthreads();
    for (int st = 128; st > 0; st >>= 1) {
        if (threadIdx.x < st) {
            rs[threadIdx.x]  += rs[threadIdx.x + st];
            rs2[threadIdx.x] += rs2[threadIdx.x + st];
        }
        __syncthreads();
    }
    if (threadIdx.x == 0) {
        const float N = 1048576.f;
        float sum = rs[0], sq = rs2[0];
        float var = (sq - sum * sum / N) / (N - 1.f);   // ddof=1
        float gs = sqrtf(var > 0.f ? var : 0.f);
        int c = bc & 3;
        float w0 = chw[0], w1 = chw[1], w2 = chw[2], w3 = chw[3];
        float mx = fmaxf(fmaxf(w0, w1), fmaxf(w2, w3));
        float e0 = __expf(w0 - mx), e1 = __expf(w1 - mx);
        float e2 = __expf(w2 - mx), e3 = __expf(w3 - mx);
        float cw = (c == 0 ? e0 : c == 1 ? e1 : c == 2 ? e2 : e3) / (e0 + e1 + e2 + e3);
        float cs = cw * gs;
        float fg = fminf(fmaxf(gs * 5.f, 0.5f), 2.f);
        float fc = fminf(fmaxf(cs * 2.f, 0.8f), 1.2f);
        stats[bc]      = 0.05f * fg * fc;   // lower
        stats[16 + bc] = 0.20f * fg * fc;   // upper
    }
}

// ---------------------------------------------------------------------------
// Pass 1: fused conv1+leaky -> conv2+leaky -> conv3 via WMMA implicit GEMM.
// Workgroup: 256 threads (8 waves), output tile 32x16.
// LDS activation layouts are CHANNEL-LAST and K is TAP-MAJOR (k=tap*Cin+cin)
// so each 8-element run of the A-fragment layout is contiguous
// (ds_load_b64/b128). Weights are stored in LDS in FRAGMENT ORDER
// ([kt][lane][16]) so B preloads are 32B contiguous per lane.
// ---------------------------------------------------------------------------
#define TW 32
#define TH 16

__global__ __launch_bounds__(256) void fused_conv_wmma(
    const float* __restrict__ x,
    const float* __restrict__ w1g, const float* __restrict__ b1g,
    const float* __restrict__ w2g, const float* __restrict__ b2g,
    const float* __restrict__ w3g, const float* __restrict__ b3g,
    float* __restrict__ fbuf)
{
    __shared__ _Float16 xs [22 * 38 * 4];    // x tile, halo 3, channel-last
    __shared__ _Float16 f1s[20 * 36 * 32];   // conv1 out (ext, halo 2), ch-last
    __shared__ _Float16 f2s[18 * 34 * 16];   // conv2 out (ext, halo 1), ch-last
    __shared__ _Float16 w1f[2 * 2 * 32 * 16]; // [kt][nt][lane][i], frag order
    __shared__ _Float16 w2f[9 * 32 * 16];     // [kt][lane][i], frag order
    __shared__ _Float16 w3f[5 * 32 * 16];     // [kt][lane][i], frag order

    const int tid  = threadIdx.x;
    const int lane = tid & 31, wave = tid >> 5;
    const int hi = lane >> 4, ln = lane & 15;
    const int tx = blockIdx.x, ty = blockIdx.y, b = blockIdx.z;
    const int gx0 = tx * TW - 3, gy0 = ty * TH - 3;

    // ---- stage x tile, channel-last, vectorized b64 store (SAME zero pad)
    for (int pix = tid; pix < 22 * 38; pix += 256) {
        int ly = pix / 38, lx = pix % 38;
        int gy = gy0 + ly, gx = gx0 + lx;
        v4h q;
        bool inb = (unsigned)gy < 1024u && (unsigned)gx < 1024u;
#pragma unroll
        for (int cin = 0; cin < 4; ++cin) {
            float v = inb ? x[(((size_t)(b * 4 + cin)) * 1024 + gy) * 1024 + gx] : 0.f;
            q[cin] = (_Float16)v;
        }
        *(v4h*)&xs[pix * 4] = q;
    }
    // ---- stage weights in FRAGMENT order; tap-major k = tap*Cin + cin
    // lane' = (idx/16)&31 ; fragment element i = idx&15 ; k = kt*32+(i>>3)*16+(lane'>>4)*8+(i&7)
    for (int idx = tid; idx < 2 * 2 * 32 * 16; idx += 256) {
        int i = idx & 15, l = (idx >> 4) & 31, nt = (idx >> 9) & 1, kt = idx >> 10;
        int k = kt * 32 + (i >> 3) * 16 + (l >> 4) * 8 + (i & 7), n = nt * 16 + (l & 15);
        float v = 0.f;
        if (k < 36) { int tap = k >> 2, cin = k & 3; v = w1g[(n * 4 + cin) * 9 + tap]; }
        w1f[idx] = (_Float16)v;
    }
    for (int idx = tid; idx < 9 * 32 * 16; idx += 256) {
        int i = idx & 15, l = (idx >> 4) & 31, kt = idx >> 9;
        int cin = (i >> 3) * 16 + (l >> 4) * 8 + (i & 7), n = l & 15;     // tap == kt for layer 2
        w2f[idx] = (_Float16)w2g[(n * 32 + cin) * 9 + kt];
    }
    for (int idx = tid; idx < 5 * 32 * 16; idx += 256) {
        int i = idx & 15, l = (idx >> 4) & 31, kt = idx >> 9;
        int k = kt * 32 + (i >> 3) * 16 + (l >> 4) * 8 + (i & 7), n = l & 15;
        float v = 0.f;
        if (k < 144 && n < 8) { int tap = k >> 4, cin = k & 15; v = w3g[(n * 16 + cin) * 9 + tap]; }
        w3f[idx] = (_Float16)v;
    }
    __syncthreads();

    // ================= Layer 1: 4->32, K=36 (pad 64, 2 k-steps), 36x20 tile
    {
        const float bias0 = b1g[ln], bias1 = b1g[16 + ln];
        v16h b1f[2][2];
#pragma unroll
        for (int kt = 0; kt < 2; ++kt)
#pragma unroll
            for (int nt = 0; nt < 2; ++nt)
                b1f[kt][nt] = *(const v16h*)&w1f[(((kt * 2 + nt) * 32) + lane) * 16];

        for (int t = wave; t < 45; t += 8) {
            v8f acc0 = {}; v8f acc1 = {};
            int p = t * 16 + ln;
            int ey = p / 36, ex = p % 36;
            const _Float16* xp = &xs[(ey * 38 + ex) * 4];
#pragma unroll
            for (int kt = 0; kt < 2; ++kt) {
                v16h a;
                // four taps per fragment: kt*8 + hi*2 + {0,1}, kt*8+4+hi*2 + {0,1}
#pragma unroll
                for (int r = 0; r < 4; ++r) {
                    int tap = kt * 8 + (r >> 1) * 4 + hi * 2 + (r & 1);
                    v4h q = {};
                    if (tap < 9) {
                        int dy = tap / 3, dx = tap % 3;
                        q = *(const v4h*)(xp + (dy * 38 + dx) * 4);
                    }
#pragma unroll
                    for (int j = 0; j < 4; ++j) a[r * 4 + j] = q[j];
                }
                acc0 = __builtin_amdgcn_wmma_f32_16x16x32_f16(false, a, false, b1f[kt][0], (short)0, acc0, false, false);
                acc1 = __builtin_amdgcn_wmma_f32_16x16x32_f16(false, a, false, b1f[kt][1], (short)0, acc1, false, false);
            }
#pragma unroll
            for (int e = 0; e < 8; ++e) {
                int pp = t * 16 + e + hi * 8;          // C row m = e + 8*hi
                int eyy = pp / 36, exx = pp % 36;
                float r0 = acc0[e] + bias0;
                float r1 = acc1[e] + bias1;
                r0 = r0 >= 0.f ? r0 : 0.2f * r0;       // leaky(0.2)
                r1 = r1 >= 0.f ? r1 : 0.2f * r1;
                f1s[(eyy * 36 + exx) * 32 + ln]      = (_Float16)r0;
                f1s[(eyy * 36 + exx) * 32 + 16 + ln] = (_Float16)r1;
            }
        }
    }
    __syncthreads();

    // ================= Layer 2: 32->16, K=288 (9 k-steps = 9 taps), 34x18
    {
        const float bias = b2g[ln];
        v16h b2f[9];
#pragma unroll
        for (int kt = 0; kt < 9; ++kt)
            b2f[kt] = *(const v16h*)&w2f[(kt * 32 + lane) * 16];

        for (int t = wave; t < 39; t += 8) {
            v8f acc = {};
            int p0 = t * 16 + ln;
            int p = p0 < 612 ? p0 : 611;               // clamp tail rows
            int ey = p / 34, ex = p % 34;
            const _Float16* f1p = &f1s[(ey * 36 + ex) * 32 + hi * 8];
#pragma unroll
            for (int kt = 0; kt < 9; ++kt) {
                const int dy = kt / 3, dx = kt % 3;
                v8h r0 = *(const v8h*)(f1p + (dy * 36 + dx) * 32);
                v8h r1 = *(const v8h*)(f1p + (dy * 36 + dx) * 32 + 16);
                v16h a;
#pragma unroll
                for (int j = 0; j < 8; ++j) { a[j] = r0[j]; a[8 + j] = r1[j]; }
                acc = __builtin_amdgcn_wmma_f32_16x16x32_f16(false, a, false, b2f[kt], (short)0, acc, false, false);
            }
#pragma unroll
            for (int e = 0; e < 8; ++e) {
                int pp = t * 16 + e + hi * 8;
                if (pp < 612) {
                    int eyy = pp / 34, exx = pp % 34;
                    float r = acc[e] + bias;
                    r = r >= 0.f ? r : 0.2f * r;
                    f2s[(eyy * 34 + exx) * 16 + ln] = (_Float16)r;
                }
            }
        }
    }
    __syncthreads();

    // ================= Layer 3: 16->8, K=144 (pad 160, 5 k-steps), 32x16
    {
        const float bias = b3g[ln & 7];
        v16h b3f[5];
#pragma unroll
        for (int kt = 0; kt < 5; ++kt)
            b3f[kt] = *(const v16h*)&w3f[(kt * 32 + lane) * 16];

        for (int t = wave; t < 32; t += 8) {
            v8f acc = {};
            int p = t * 16 + ln;
            int ey = p >> 5, ex = p & 31;
            const _Float16* f2p = &f2s[(ey * 34 + ex) * 16 + hi * 8];
#pragma unroll
            for (int kt = 0; kt < 5; ++kt) {
                // run1: tap 2kt, run2: tap 2kt+1 (tap 9 is K-padding -> zero)
                const int t0 = 2 * kt, t1 = 2 * kt + 1;
                v8h r0 = *(const v8h*)(f2p + ((t0 / 3) * 34 + (t0 % 3)) * 16);
                v8h r1 = {};
                if (t1 < 9)
                    r1 = *(const v8h*)(f2p + ((t1 / 3) * 34 + (t1 % 3)) * 16);
                v16h a;
#pragma unroll
                for (int j = 0; j < 8; ++j) { a[j] = r0[j]; a[8 + j] = r1[j]; }
                acc = __builtin_amdgcn_wmma_f32_16x16x32_f16(false, a, false, b3f[kt], (short)0, acc, false, false);
            }
            if (ln < 8) {                               // only 8 real out channels
#pragma unroll
                for (int e = 0; e < 8; ++e) {
                    int pp = t * 16 + e + hi * 8;
                    int yy = pp >> 5, xx = pp & 31;
                    float r = acc[e] + bias;            // no activation on conv3
                    fbuf[(((size_t)(b * 8 + ln)) * 1024 + ty * TH + yy) * 1024 + tx * TW + xx] = r;
                }
            }
        }
    }
}

// ---------------------------------------------------------------------------
// Pass 2: multi-window std maps (reflect pad, separable box sums) + fusion
//         + adaptive threshold + sigmoid. f-tile staging uses CDNA5 async
//         global->LDS copies (ASYNCcnt) when the toolchain exposes them.
// ---------------------------------------------------------------------------
__device__ __forceinline__ int refl(int i, int n) {
    if (i < 0) i = -i;
    if (i >= n) i = 2 * n - 2 - i;
    return i;
}

__global__ __launch_bounds__(256) void stdmap_kernel(
    const float* __restrict__ fbuf, const float* __restrict__ fusion_w,
    const float* __restrict__ fusion_b, const float* __restrict__ stats,
    float* __restrict__ out)
{
    __shared__ float ft [46 * 46];
    __shared__ float hs [46 * 32];
    __shared__ float hs2[46 * 32];
    const int tid = threadIdx.x;
    const int tx = blockIdx.x, ty = blockIdx.y, b = blockIdx.z;
    float sstd[3][4] = {};
    const int wins[3] = {5, 9, 15};

    for (int c = 0; c < 8; ++c) {
        __syncthreads();                    // previous-iter readers done
        for (int idx = tid; idx < 46 * 46; idx += 256) {
            int ly = idx / 46, lx = idx % 46;
            int gy = refl(ty * 32 - 7 + ly, 1024);
            int gx = refl(tx * 32 - 7 + lx, 1024);
            const float* gp = &fbuf[(((size_t)(b * 8 + c)) * 1024 + gy) * 1024 + gx];
#if HAS_ASYNC_LDS
            __builtin_amdgcn_global_load_async_to_lds_b32(
                (glob_i32p)(void*)gp, (lds_i32p)(void*)&ft[idx], 0, 0);
#else
            ft[idx] = *gp;
#endif
        }
#if HAS_ASYNC_LDS
        __builtin_amdgcn_s_wait_asynccnt(0);
#endif
        __syncthreads();
        for (int w = 0; w < 3; ++w) {
            int win = wins[w], pad = win >> 1;
            // horizontal box sums of f and f^2
            for (int idx = tid; idx < 46 * 32; idx += 256) {
                int ry = idx >> 5, cx = idx & 31;
                float s = 0.f, s2 = 0.f;
                for (int d = 0; d < win; ++d) {
                    float v = ft[ry * 46 + cx + 7 - pad + d];
                    s += v; s2 += v * v;
                }
                hs[idx] = s; hs2[idx] = s2;
            }
            __syncthreads();
            // vertical accumulation -> std, averaged over 8 channels
            float inv = 1.f / (float)(win * win);
#pragma unroll
            for (int i = 0; i < 4; ++i) {
                int pid = tid + 256 * i;
                int py = pid >> 5, px = pid & 31;
                float m = 0.f, m2 = 0.f;
                for (int d = 0; d < win; ++d) {
                    int r = py + 7 - pad + d;
                    m  += hs [r * 32 + px];
                    m2 += hs2[r * 32 + px];
                }
                float mean = m * inv, mean2 = m2 * inv;
                float var = mean2 - mean * mean;
                var = var > 0.f ? var : 0.f;
                sstd[w][i] += sqrtf(var + 1e-8f) * 0.125f;
            }
            __syncthreads();
        }
    }

    float fw0 = fusion_w[0], fw1 = fusion_w[1], fw2 = fusion_w[2], fb = fusion_b[0];
#pragma unroll
    for (int i = 0; i < 4; ++i) {
        int pid = tid + 256 * i;
        int py = pid >> 5, px = pid & 31;
        int gy = ty * 32 + py, gx = tx * 32 + px;
        float fused = fw0 * sstd[0][i] + fw1 * sstd[1][i] + fw2 * sstd[2][i] + fb;
        for (int oc = 0; oc < 4; ++oc) {
            float lo = stats[b * 4 + oc];
            float up = stats[16 + b * 4 + oc];
            float ns = (fused - lo) / (up - lo);
            ns = ns < 0.f ? 0.f : (ns > 1.f ? 1.f : ns);
            out[(((size_t)(b * 4 + oc)) * 1024 + gy) * 1024 + gx] =
                1.f / (1.f + __expf(3.f - 6.f * ns));
        }
    }
}

// ---------------------------------------------------------------------------
extern "C" void kernel_launch(void* const* d_in, const int* in_sizes, int n_in,
                              void* d_out, int out_size, void* d_ws, size_t ws_size,
                              hipStream_t stream) {
    const float* x    = (const float*)d_in[0];
    const float* c1w  = (const float*)d_in[1];
    const float* c1b  = (const float*)d_in[2];
    const float* c2w  = (const float*)d_in[3];
    const float* c2b  = (const float*)d_in[4];
    const float* c3w  = (const float*)d_in[5];
    const float* c3b  = (const float*)d_in[6];
    const float* chw  = (const float*)d_in[7];
    const float* fw   = (const float*)d_in[8];
    const float* fb   = (const float*)d_in[9];
    float* stats = (float*)d_ws;                         // 32 floats
    float* fbuf  = (float*)((char*)d_ws + 256);          // 4*8*1024*1024 f32

    stats_kernel<<<16, 256, 0, stream>>>(x, chw, stats);
    fused_conv_wmma<<<dim3(32, 64, 4), 256, 0, stream>>>(
        x, c1w, c1b, c2w, c2b, c3w, c3b, fbuf);
    stdmap_kernel<<<dim3(32, 32, 4), 256, 0, stream>>>(
        fbuf, fw, fb, stats, (float*)d_out);
}
